// MultimodalFusion_75900662055429
// MI455X (gfx1250) — matrix operand order repaired and checked by reading, not compiled, not run on hardware
//
#include <hip/hip_runtime.h>
#include <hip/hip_bf16.h>
#include <math.h>

typedef __attribute__((ext_vector_type(16))) __bf16 v16bf;
typedef __attribute__((ext_vector_type(8)))  __bf16 v8bf;
typedef __attribute__((ext_vector_type(8)))  float  v8f;

union ABf { v16bf v; v8bf h[2]; };

__device__ __forceinline__ float gelu_exact(float x) {
  return 0.5f * x * (1.0f + erff(x * 0.70710678118654752440f));
}

// ---------------- mean pooling: [B,S,D] f32 -> [B,D] bf16 ----------------
__global__ void mean_pool_kernel(const float* __restrict__ in, __bf16* __restrict__ out,
                                 int S, int D) {
  int b = blockIdx.x;
  int c = threadIdx.x * 4;
  if (c >= D) return;
  const float4* p = reinterpret_cast<const float4*>(in + (size_t)b * S * D + c);
  int strideF4 = D >> 2;
  float4 s0 = make_float4(0.f, 0.f, 0.f, 0.f);
  float4 s1 = make_float4(0.f, 0.f, 0.f, 0.f);
  for (int t = 0; t < S; t += 2) {   // S is even for all three modalities
    float4 x0 = p[(size_t)t * strideF4];
    float4 x1 = p[(size_t)(t + 1) * strideF4];
    s0.x += x0.x; s0.y += x0.y; s0.z += x0.z; s0.w += x0.w;
    s1.x += x1.x; s1.y += x1.y; s1.z += x1.z; s1.w += x1.w;
  }
  float inv = 1.0f / (float)S;
  __bf16* o = out + (size_t)b * D + c;
  o[0] = (__bf16)((s0.x + s1.x) * inv);
  o[1] = (__bf16)((s0.y + s1.y) * inv);
  o[2] = (__bf16)((s0.z + s1.z) * inv);
  o[3] = (__bf16)((s0.w + s1.w) * inv);
}

// ---------------- f32 -> bf16 convert ----------------
__global__ void cvt_bf16_kernel(const float* __restrict__ in, __bf16* __restrict__ out, int n) {
  int i = blockIdx.x * blockDim.x + threadIdx.x;
  int stride = gridDim.x * blockDim.x;
  for (; i < n; i += stride) out[i] = (__bf16)in[i];
}

// ---------------- WMMA bf16 GEMM: C[M,N] = A[M,K] * W[N,K]^T + bias ----------------
// One wave computes a 32x64 tile: 2 M-fragments x 4 N-fragments = 8 accumulators,
// reusing each A fragment across 4 WMMAs and each B fragment across 2.
// Epilogue: bias, optional residual, optional exact gelu, f32 C + optional bf16 shadow.
__global__ __launch_bounds__(32)
void wmma_gemm_bf16(const __bf16* __restrict__ A, int lda,
                    const __bf16* __restrict__ W, int ldw,
                    const float* __restrict__ bias,
                    const float* __restrict__ resid, int ldr,
                    float* __restrict__ C, int ldc,
                    __bf16* __restrict__ Cbf, int ldcb,
                    int K, int do_gelu) {
  const int lane = threadIdx.x;      // 0..31, wave32
  const int mt = blockIdx.y * 32;
  const int nt = blockIdx.x * 64;
  const int l15 = lane & 15;
  const int hi  = lane >> 4;         // 0: lanes 0-15, 1: lanes 16-31
  // A fragment (16x32 bf16): lane<16 gets K = k0+{0..7, 16..23}; lane>=16 offset +8
  const __bf16* aRow0 = A + (size_t)(mt + l15) * lda + hi * 8;
  const __bf16* aRow1 = aRow0 + (size_t)16 * lda;
  // B fragment (32x16 bf16) from W rows [N,K]: lane<16 K=k0+{0..7, 16..23}; lane>=16 offset +8
  const __bf16* wRow0 = W + (size_t)(nt + l15) * ldw + hi * 8;
  v8f acc[2][4] = {};
  for (int k0 = 0; k0 < K; k0 += 32) {
    ABf af[2];
    af[0].h[0] = *reinterpret_cast<const v8bf*>(aRow0 + k0);
    af[0].h[1] = *reinterpret_cast<const v8bf*>(aRow0 + k0 + 16);
    af[1].h[0] = *reinterpret_cast<const v8bf*>(aRow1 + k0);
    af[1].h[1] = *reinterpret_cast<const v8bf*>(aRow1 + k0 + 16);
    ABf bfr[4];
#pragma unroll
    for (int j = 0; j < 4; ++j) {
      const __bf16* wr = wRow0 + (size_t)(j * 16) * ldw + k0;
      bfr[j].h[0] = *reinterpret_cast<const v8bf*>(wr);
      bfr[j].h[1] = *reinterpret_cast<const v8bf*>(wr + 16);
    }
#pragma unroll
    for (int i = 0; i < 2; ++i)
#pragma unroll
      for (int j = 0; j < 4; ++j)
        acc[i][j] = __builtin_amdgcn_wmma_f32_16x16x32_bf16(
            false, af[i].v, false, bfr[j].v, (short)0, acc[i][j], false, false);
  }
#pragma unroll
  for (int j = 0; j < 4; ++j) {
    const int n = nt + j * 16 + l15;
    const float bval = bias ? bias[n] : 0.0f;
#pragma unroll
    for (int i = 0; i < 2; ++i) {
#pragma unroll
      for (int r = 0; r < 8; ++r) {
        int m = mt + i * 16 + r + hi * 8;  // C/D: VGPR r -> M=r (lanes 0-15), r+8 (lanes 16-31)
        float v = acc[i][j][r] + bval;
        if (resid) v += resid[(size_t)m * ldr + n];
        if (do_gelu) v = gelu_exact(v);
        C[(size_t)m * ldc + n] = v;
        if (Cbf) Cbf[(size_t)m * ldcb + n] = (__bf16)v;
      }
    }
  }
}

// ---------------- LayerNorm over rows of width D ----------------
__global__ void layernorm_kernel(const float* __restrict__ x, int ldx,
                                 const float* __restrict__ g, const float* __restrict__ b,
                                 float* __restrict__ yf, int ldy,
                                 __bf16* __restrict__ yb, int ldyb, int D) {
  __shared__ float ssum[256], ssq[256];
  int row = blockIdx.x;
  const float* xr = x + (size_t)row * ldx;
  float s = 0.f, sq = 0.f;
  for (int c = threadIdx.x; c < D; c += blockDim.x) {
    float v = xr[c]; s += v; sq += v * v;
  }
  ssum[threadIdx.x] = s; ssq[threadIdx.x] = sq;
  __syncthreads();
  for (int off = blockDim.x >> 1; off > 0; off >>= 1) {
    if ((int)threadIdx.x < off) {
      ssum[threadIdx.x] += ssum[threadIdx.x + off];
      ssq[threadIdx.x]  += ssq[threadIdx.x + off];
    }
    __syncthreads();
  }
  float mu   = ssum[0] / (float)D;
  float var  = ssq[0] / (float)D - mu * mu;
  float rstd = rsqrtf(var + 1e-5f);
  for (int c = threadIdx.x; c < D; c += blockDim.x) {
    float v = (xr[c] - mu) * rstd * g[c] + b[c];
    if (yf) yf[(size_t)row * ldy + c] = v;
    if (yb) yb[(size_t)row * ldyb + c] = (__bf16)v;
  }
}

// ---------------- S=3 multi-head attention (8 heads, dh=64) ----------------
// qkv rows: b*3+s, 1536 cols [q|k|v]; one wave per (b, head), 2 dims per lane.
__global__ void attn3_kernel(const float* __restrict__ qkv, __bf16* __restrict__ obf) {
  int b = blockIdx.x;
  int h = threadIdx.x >> 5;
  int lane = threadIdx.x & 31;
  int d0 = lane * 2;
  float q[3][2], k[3][2], v[3][2];
  for (int s = 0; s < 3; ++s) {
    const float* row = qkv + (size_t)(b * 3 + s) * 1536 + h * 64 + d0;
    q[s][0] = row[0];    q[s][1] = row[1];
    k[s][0] = row[512];  k[s][1] = row[513];
    v[s][0] = row[1024]; v[s][1] = row[1025];
  }
  float sc[3][3];
  for (int qs = 0; qs < 3; ++qs)
    for (int ks = 0; ks < 3; ++ks) {
      float p = q[qs][0] * k[ks][0] + q[qs][1] * k[ks][1];
      for (int off = 16; off > 0; off >>= 1) p += __shfl_xor(p, off, 32);
      sc[qs][ks] = p * 0.125f;   // 1/sqrt(64)
    }
  for (int qs = 0; qs < 3; ++qs) {
    float m = fmaxf(sc[qs][0], fmaxf(sc[qs][1], sc[qs][2]));
    float e0 = expf(sc[qs][0] - m), e1 = expf(sc[qs][1] - m), e2 = expf(sc[qs][2] - m);
    float inv = 1.0f / (e0 + e1 + e2);
    float o0 = (e0 * v[0][0] + e1 * v[1][0] + e2 * v[2][0]) * inv;
    float o1 = (e0 * v[0][1] + e1 * v[1][1] + e2 * v[2][1]) * inv;
    __bf16* orow = obf + (size_t)(b * 3 + qs) * 512 + h * 64 + d0;
    orow[0] = (__bf16)o0; orow[1] = (__bf16)o1;
  }
}

// ---------------- gate logits + softmax: flat[B,1536] @ gate_w[3,1536]^T ----------------
__global__ __launch_bounds__(32)
void gates_kernel(const float* __restrict__ flat, const float* __restrict__ gw,
                  const float* __restrict__ gb, float* __restrict__ gates) {
  int b = blockIdx.x;
  int lane = threadIdx.x;
  float a0 = 0.f, a1 = 0.f, a2 = 0.f;
  const float* xr = flat + (size_t)b * 1536;
  for (int c = lane; c < 1536; c += 32) {
    float x = xr[c];
    a0 += x * gw[c];
    a1 += x * gw[1536 + c];
    a2 += x * gw[3072 + c];
  }
  for (int off = 16; off > 0; off >>= 1) {
    a0 += __shfl_xor(a0, off, 32);
    a1 += __shfl_xor(a1, off, 32);
    a2 += __shfl_xor(a2, off, 32);
  }
  a0 += gb[0]; a1 += gb[1]; a2 += gb[2];
  float m = fmaxf(a0, fmaxf(a1, a2));
  float e0 = expf(a0 - m), e1 = expf(a1 - m), e2 = expf(a2 - m);
  float inv = 1.0f / (e0 + e1 + e2);
  if (lane == 0) {
    gates[b * 3 + 0] = e0 * inv;
    gates[b * 3 + 1] = e1 * inv;
    gates[b * 3 + 2] = e2 * inv;
  }
}

// ---------------- fused = sum_n g[n]*refined[b,n,:] + residual + ffn_out ----------------
__global__ void fuse_kernel(const float* __restrict__ refined, const float* __restrict__ gates,
                            const float* __restrict__ residual, const float* __restrict__ ffn_out,
                            float* __restrict__ fused) {
  int idx = blockIdx.x * blockDim.x + threadIdx.x;   // B*512
  int b = idx >> 9, d = idx & 511;
  const float* rr = refined + (size_t)b * 1536;
  float g0 = gates[b * 3], g1 = gates[b * 3 + 1], g2 = gates[b * 3 + 2];
  fused[idx] = g0 * rr[d] + g1 * rr[512 + d] + g2 * rr[1024 + d] + residual[idx] + ffn_out[idx];
}

// ---------------- final [B,64] @ op_w3[10,64]^T + b3 ----------------
__global__ void head_final_kernel(const float* __restrict__ h2, const float* __restrict__ w3,
                                  const float* __restrict__ b3, float* __restrict__ out) {
  int b = blockIdx.x;
  int j = threadIdx.x;
  if (j >= 10) return;
  const float* x = h2 + (size_t)b * 64;
  const float* w = w3 + (size_t)j * 64;
  float s = b3[j];
#pragma unroll
  for (int kk = 0; kk < 64; ++kk) s += x[kk] * w[kk];
  out[b * 10 + j] = s;
}

static inline size_t align256(size_t x) { return (x + 255) & ~(size_t)255; }

extern "C" void kernel_launch(void* const* d_in, const int* in_sizes, int n_in,
                              void* d_out, int out_size, void* d_ws, size_t ws_size,
                              hipStream_t stream) {
  (void)in_sizes; (void)n_in; (void)out_size; (void)ws_size;
  const int B = 1024;
  const float* vision   = (const float*)d_in[0];
  const float* audio    = (const float*)d_in[1];
  const float* text     = (const float*)d_in[2];
  const float* vp_w     = (const float*)d_in[3];
  const float* vp_b     = (const float*)d_in[4];
  const float* ap_w     = (const float*)d_in[5];
  const float* ap_b     = (const float*)d_in[6];
  const float* tp_w     = (const float*)d_in[7];
  const float* tp_b     = (const float*)d_in[8];
  const float* vln_g    = (const float*)d_in[9];
  const float* vln_b    = (const float*)d_in[10];
  const float* aln_g    = (const float*)d_in[11];
  const float* aln_b    = (const float*)d_in[12];
  const float* tln_g    = (const float*)d_in[13];
  const float* tln_b    = (const float*)d_in[14];
  const float* ca_in_w  = (const float*)d_in[15];
  const float* ca_in_b  = (const float*)d_in[16];
  const float* ca_out_w = (const float*)d_in[17];
  const float* ca_out_b = (const float*)d_in[18];
  const float* sa_in_w  = (const float*)d_in[19];
  const float* sa_in_b  = (const float*)d_in[20];
  const float* sa_out_w = (const float*)d_in[21];
  const float* sa_out_b = (const float*)d_in[22];
  const float* ffn_w1   = (const float*)d_in[23];
  const float* ffn_b1   = (const float*)d_in[24];
  const float* ffn_w2   = (const float*)d_in[25];
  const float* ffn_b2   = (const float*)d_in[26];
  const float* gate_w   = (const float*)d_in[27];
  const float* gate_b   = (const float*)d_in[28];
  const float* res_w    = (const float*)d_in[29];
  const float* res_b    = (const float*)d_in[30];
  const float* oln_g    = (const float*)d_in[31];
  const float* oln_b    = (const float*)d_in[32];
  const float* op_w1    = (const float*)d_in[33];
  const float* op_b1    = (const float*)d_in[34];
  const float* op_w2    = (const float*)d_in[35];
  const float* op_b2    = (const float*)d_in[36];
  const float* op_w3    = (const float*)d_in[37];
  const float* op_b3    = (const float*)d_in[38];

  char* ws = (char*)d_ws;
  size_t off = 0;
  auto alloc = [&](size_t bytes) -> void* {
    void* p = ws + off;
    off = align256(off + bytes);
    return p;
  };

  // bf16 copies of WMMA weights
  __bf16* w_vp   = (__bf16*)alloc((size_t)512 * 1024 * 2);
  __bf16* w_ap   = (__bf16*)alloc((size_t)512 * 512 * 2);
  __bf16* w_tp   = (__bf16*)alloc((size_t)512 * 768 * 2);
  __bf16* w_cain = (__bf16*)alloc((size_t)1536 * 512 * 2);
  __bf16* w_caout= (__bf16*)alloc((size_t)512 * 512 * 2);
  __bf16* w_sain = (__bf16*)alloc((size_t)1536 * 512 * 2);
  __bf16* w_saout= (__bf16*)alloc((size_t)512 * 512 * 2);
  __bf16* w_ffn1 = (__bf16*)alloc((size_t)1024 * 1536 * 2);
  __bf16* w_ffn2 = (__bf16*)alloc((size_t)512 * 1024 * 2);
  __bf16* w_res  = (__bf16*)alloc((size_t)512 * 1536 * 2);
  __bf16* w_op1  = (__bf16*)alloc((size_t)256 * 512 * 2);
  __bf16* w_op2  = (__bf16*)alloc((size_t)64 * 256 * 2);
  // activations
  __bf16* pool_v = (__bf16*)alloc((size_t)B * 1024 * 2);
  __bf16* pool_a = (__bf16*)alloc((size_t)B * 512 * 2);
  __bf16* pool_t = (__bf16*)alloc((size_t)B * 768 * 2);
  float*  stacked    = (float*)alloc((size_t)B * 1536 * 4);   // [B,3,512]
  __bf16* stacked_bf = (__bf16*)alloc((size_t)B * 1536 * 2);
  float*  qkv        = (float*)alloc((size_t)B * 3 * 1536 * 4);
  __bf16* attnO      = (__bf16*)alloc((size_t)B * 3 * 512 * 2);
  float*  attn_out   = (float*)alloc((size_t)B * 3 * 512 * 4);
  __bf16* attn_out_bf= (__bf16*)alloc((size_t)B * 3 * 512 * 2);
  float*  refined    = (float*)alloc((size_t)B * 3 * 512 * 4); // == flat [B,1536]
  __bf16* refined_bf = (__bf16*)alloc((size_t)B * 3 * 512 * 2);
  float*  ffn_h      = (float*)alloc((size_t)B * 1024 * 4);
  __bf16* ffn_h_bf   = (__bf16*)alloc((size_t)B * 1024 * 2);
  float*  ffn_out    = (float*)alloc((size_t)B * 512 * 4);
  float*  residualf  = (float*)alloc((size_t)B * 512 * 4);
  float*  gatesf     = (float*)alloc((size_t)B * 3 * 4);
  float*  fused      = (float*)alloc((size_t)B * 512 * 4);
  __bf16* fused_bf   = (__bf16*)alloc((size_t)B * 512 * 2);
  float*  h1f        = (float*)alloc((size_t)B * 256 * 4);
  __bf16* h1_bf      = (__bf16*)alloc((size_t)B * 256 * 2);
  float*  h2f        = (float*)alloc((size_t)B * 64 * 4);

  auto cvt = [&](const float* src, __bf16* dst, int n) {
    int blocks = (n + 255) / 256;
    if (blocks > 2048) blocks = 2048;
    cvt_bf16_kernel<<<blocks, 256, 0, stream>>>(src, dst, n);
  };
  auto gemm = [&](const __bf16* A, int lda, const __bf16* W, int ldw, const float* bias,
                  const float* resid, int ldr, float* C, int ldc, __bf16* Cbf, int ldcb,
                  int M, int N, int K, int do_gelu) {
    wmma_gemm_bf16<<<dim3(N / 64, M / 32), 32, 0, stream>>>(
        A, lda, W, ldw, bias, resid, ldr, C, ldc, Cbf, ldcb, K, do_gelu);
  };

  // 0) weight conversions
  cvt(vp_w, w_vp, 512 * 1024);
  cvt(ap_w, w_ap, 512 * 512);
  cvt(tp_w, w_tp, 512 * 768);
  cvt(ca_in_w, w_cain, 1536 * 512);
  cvt(ca_out_w, w_caout, 512 * 512);
  cvt(sa_in_w, w_sain, 1536 * 512);
  cvt(sa_out_w, w_saout, 512 * 512);
  cvt(ffn_w1, w_ffn1, 1024 * 1536);
  cvt(ffn_w2, w_ffn2, 512 * 1024);
  cvt(res_w, w_res, 512 * 1536);
  cvt(op_w1, w_op1, 256 * 512);
  cvt(op_w2, w_op2, 64 * 256);

  // 1) mean pools (the memory-bound stage: 1.76 GB streamed)
  mean_pool_kernel<<<B, 256, 0, stream>>>(vision, pool_v, 196, 1024);
  mean_pool_kernel<<<B, 128, 0, stream>>>(audio,  pool_a, 256, 512);
  mean_pool_kernel<<<B, 192, 0, stream>>>(text,   pool_t, 128, 768);

  // 2) modality projections into stacked[B,3,512]
  gemm(pool_v, 1024, w_vp, 1024, vp_b, nullptr, 0, stacked + 0,    1536, nullptr, 0, B, 512, 1024, 0);
  gemm(pool_a, 512,  w_ap, 512,  ap_b, nullptr, 0, stacked + 512,  1536, nullptr, 0, B, 512, 512,  0);
  gemm(pool_t, 768,  w_tp, 768,  tp_b, nullptr, 0, stacked + 1024, 1536, nullptr, 0, B, 512, 768,  0);

  // 3) per-modality layernorms (in place, bf16 shadow)
  layernorm_kernel<<<B, 256, 0, stream>>>(stacked + 0,    1536, vln_g, vln_b, stacked + 0,    1536, stacked_bf + 0,    1536, 512);
  layernorm_kernel<<<B, 256, 0, stream>>>(stacked + 512,  1536, aln_g, aln_b, stacked + 512,  1536, stacked_bf + 512,  1536, 512);
  layernorm_kernel<<<B, 256, 0, stream>>>(stacked + 1024, 1536, tln_g, tln_b, stacked + 1024, 1536, stacked_bf + 1024, 1536, 512);

  // 4) cross-attention MHA
  gemm(stacked_bf, 512, w_cain, 512, ca_in_b, nullptr, 0, qkv, 1536, nullptr, 0, B * 3, 1536, 512, 0);
  attn3_kernel<<<B, 256, 0, stream>>>(qkv, attnO);
  gemm(attnO, 512, w_caout, 512, ca_out_b, nullptr, 0, attn_out, 512, attn_out_bf, 512, B * 3, 512, 512, 0);

  // 5) self-attention MHA + residual (refined = mha2(attn_out) + attn_out)
  gemm(attn_out_bf, 512, w_sain, 512, sa_in_b, nullptr, 0, qkv, 1536, nullptr, 0, B * 3, 1536, 512, 0);
  attn3_kernel<<<B, 256, 0, stream>>>(qkv, attnO);
  gemm(attnO, 512, w_saout, 512, sa_out_b, attn_out, 512, refined, 512, refined_bf, 512, B * 3, 512, 512, 0);

  // 6) FFN on flat [B,1536]
  gemm(refined_bf, 1536, w_ffn1, 1536, ffn_b1, nullptr, 0, ffn_h, 1024, ffn_h_bf, 1024, B, 1024, 1536, 1);
  gemm(ffn_h_bf, 1024, w_ffn2, 1024, ffn_b2, nullptr, 0, ffn_out, 512, nullptr, 0, B, 512, 1024, 0);

  // 7) residual proj + gates + fuse
  gemm(refined_bf, 1536, w_res, 1536, res_b, nullptr, 0, residualf, 512, nullptr, 0, B, 512, 1536, 0);
  gates_kernel<<<B, 32, 0, stream>>>(refined, gate_w, gate_b, gatesf);
  fuse_kernel<<<(B * 512) / 256, 256, 0, stream>>>(refined, gatesf, residualf, ffn_out, fused);

  // 8) output head
  layernorm_kernel<<<B, 256, 0, stream>>>(fused, 512, oln_g, oln_b, nullptr, 0, fused_bf, 512, 512);
  gemm(fused_bf, 512, w_op1, 512, op_b1, nullptr, 0, h1f, 256, h1_bf, 256, B, 256, 512, 1);
  gemm(h1_bf, 256, w_op2, 256, op_b2, nullptr, 0, h2f, 64, nullptr, 0, B, 64, 256, 1);
  head_final_kernel<<<B, 16, 0, stream>>>(h2f, op_w3, op_b3, (float*)d_out);
}
